// NeRF_29884382446000
// MI455X (gfx1250) — matrix operand order repaired and checked by reading, not compiled, not run on hardware
//
#include <hip/hip_runtime.h>
#include <math.h>

#define TSZ 524288          // T = 2^19 hash table entries per level

typedef __attribute__((ext_vector_type(16))) _Float16 v16h;
typedef __attribute__((ext_vector_type(8)))  _Float16 v8h;
typedef __attribute__((ext_vector_type(8)))  float    v8f;

struct ResT { float r[16]; };

// ---- weight fragment tile ids (pre-swizzled B layout, 512 f16 per tile) ----
// sw0: tiles 0..3   (kc=0, nt=0..3)        [32x64]
// sw1: tiles 4..5   (kc=0..1, nt=0)        [64x16]
// cw0: tiles 6..17  (6+kc*4+nt, kc=0..2)   [96(79)x64]
// cw1: tiles 18..25 (18+kc*4+nt, kc=0..1)  [64x64]
// cw2: tiles 26..27 (kc=0..1, nt=0)        [64x16(3)]
#define T_SW0(nt)     (nt)
#define T_SW1(kc)     (4 + (kc))
#define T_CW0(kc,nt)  (6 + (kc)*4 + (nt))
#define T_CW1(kc,nt)  (18 + (kc)*4 + (nt))
#define T_CW2(kc)     (26 + (kc))
#define N_TILES 28

// ---- WMMA fragment helpers (layouts per CDNA5 ISA 7.12.2) -----------------

// A-matrix 16x32 f16, row-major staging buffer. Per lane: two contiguous
// 16-byte chunks: k = 8*half + kc*32 + {0..7} and +16.
__device__ __forceinline__ v16h load_A16(const _Float16* buf, int stride,
                                         int kchunk, int lane) {
  const int r = lane & 15, half = lane >> 4;
  const _Float16* p = buf + r * stride + 8 * half + kchunk * 32;
  const v8h lo = *(const v8h*)(p);
  const v8h hi = *(const v8h*)(p + 16);
  return __builtin_shufflevector(lo, hi, 0, 1, 2, 3, 4, 5, 6, 7,
                                         8, 9, 10, 11, 12, 13, 14, 15);
}

// B fragment from pre-swizzled LDS tile: lane's 16 f16 are contiguous.
__device__ __forceinline__ v16h load_Bfrag(const _Float16* wfrag, int tile,
                                           int lane) {
  return *(const v16h*)(wfrag + tile * 512 + lane * 16);
}

__device__ __forceinline__ v8f wmma16(v16h a, v16h b, v8f c) {
  return __builtin_amdgcn_wmma_f32_16x16x32_f16(false, a, false, b,
                                                (short)0, c, false, false);
}

// D 16x16 f32: lane holds column n=lane&15, vgpr j -> row m=j+8*(lane>>4).
__device__ __forceinline__ void store_relu(v8f dacc, _Float16* hbuf, int stride,
                                           int ntile, const float* bias, int lane) {
  const int n    = ntile * 16 + (lane & 15);
  const int half = lane >> 4;
  const float bb = bias[n];
#pragma unroll
  for (int j = 0; j < 8; ++j) {
    float v = dacc[j] + bb;
    v = fmaxf(v, 0.0f);
    hbuf[(j + 8 * half) * stride + n] = (_Float16)v;
  }
}

// ---- fused NeRF forward ----------------------------------------------------
// block = 256 threads = 8 waves; each wave handles 16 points (128 pts/block).

__global__ __launch_bounds__(256) void nerf_fused(
    const float* __restrict__ x, const float* __restrict__ dvec,
    const int* __restrict__ app_idx, const float* __restrict__ htab,
    const float* __restrict__ atab,
    const float* __restrict__ sw0, const float* __restrict__ sb0,
    const float* __restrict__ sw1, const float* __restrict__ sb1,
    const float* __restrict__ cw0, const float* __restrict__ cb0,
    const float* __restrict__ cw1, const float* __restrict__ cb1,
    const float* __restrict__ cw2, const float* __restrict__ cb2,
    float* __restrict__ out, ResT res)
{
  __shared__ __align__(32) _Float16 wfrag[N_TILES * 512]; // B fragments
  __shared__ float    s_bias[224];                        // sb0|sb1|cb0|cb1|cb2
  __shared__ __align__(32) _Float16 stage[8 * 16 * 96];   // per-wave staging

  const int tid  = threadIdx.x;
  const int wave = tid >> 5;
  const int lane = tid & 31;
  const int p0   = blockIdx.x * 128 + wave * 16;
  _Float16* st   = stage + wave * (16 * 96);

  // ---- Phase A: swizzle weights (fp32 -> fp16) into B-fragment layout.
  // Fragment element e of lane l in a (kc) tile corresponds to
  // k = kc*32 + 8*(l>>4) + e (e<8), kc*32 + 16 + 8*(l>>4) + (e-8) (e>=8), n = ntile*16 + (l&15).
  for (int i = tid; i < N_TILES * 512; i += 256) {
    const int tile = i >> 9;
    const int r    = i & 511;
    const int ln   = r >> 4;
    const int e    = r & 15;
    const int k_local = (e < 8) ? 8 * (ln >> 4) + e : 16 + 8 * (ln >> 4) + (e - 8);   // 0..31 within K-chunk
    const int n_local = ln & 15;
    float val;
    if (tile < 4) {                                     // sw0 [32][64]
      val = sw0[k_local * 64 + (tile * 16 + n_local)];
    } else if (tile < 6) {                              // sw1 [64][16]
      val = sw1[((tile - 4) * 32 + k_local) * 16 + n_local];
    } else if (tile < 18) {                             // cw0 [79][64] K->96
      const int t = tile - 6, kc = t >> 2, nt = t & 3;
      const int k = kc * 32 + k_local;
      val = (k < 79) ? cw0[k * 64 + (nt * 16 + n_local)] : 0.0f;
    } else if (tile < 26) {                             // cw1 [64][64]
      const int t = tile - 18, kc = t >> 2, nt = t & 3;
      val = cw1[(kc * 32 + k_local) * 64 + (nt * 16 + n_local)];
    } else {                                            // cw2 [64][3] N->16
      const int k = (tile - 26) * 32 + k_local;
      val = (n_local < 3) ? cw2[k * 3 + n_local] : 0.0f;
    }
    wfrag[i] = (_Float16)val;
  }
  if (tid < 224) {
    float v; const int i = tid;
    if      (i <  64) v = sb0[i];
    else if (i <  80) v = sb1[i - 64];
    else if (i < 144) v = cb0[i - 80];
    else if (i < 208) v = cb1[i - 144];
    else              v = (i - 208 < 3) ? cb2[i - 208] : 0.0f; // cb2 padded N->16
    s_bias[i] = v;
  }
  __syncthreads();
  const float* SB0 = s_bias;
  const float* SB1 = s_bias + 64;
  const float* CB0 = s_bias + 80;
  const float* CB1 = s_bias + 144;
  const float* CB2 = s_bias + 208;

  // ---- Phase B: multi-res hash encoding -> enc tile [16][32] f16 in st
  // lane&15 -> point, lane>>4 -> level half (8 levels each).
  {
    const int pt = lane & 15;
    const int p  = p0 + pt;
    const int lo = (lane >> 4) * 8;
    const float px = (x[p * 3 + 0] + 1.0f) * 0.5f;
    const float py = (x[p * 3 + 1] + 1.0f) * 0.5f;
    const float pz = (x[p * 3 + 2] + 1.0f) * 0.5f;
#pragma unroll
    for (int li = 0; li < 8; ++li) {
      const int l = lo + li;
      const float rs = res.r[l];
      const float sx = px * rs, sy = py * rs, sz = pz * rs;
      const float fx = floorf(sx), fy = floorf(sy), fz = floorf(sz);
      const float wx = sx - fx, wy = sy - fy, wz = sz - fz;
      const unsigned cx = (unsigned)fx, cy = (unsigned)fy, cz = (unsigned)fz;
      const float2* tl = (const float2*)(htab + ((size_t)l << 20)); // l*T*2
      float a0 = 0.0f, a1 = 0.0f;
#pragma unroll
      for (int c8 = 0; c8 < 8; ++c8) {
        const unsigned ox = (c8 >> 2) & 1u, oy = (c8 >> 1) & 1u, oz = c8 & 1u;
        const unsigned h = (cx + ox) ^ ((cy + oy) * 2654435761u)
                                     ^ ((cz + oz) * 805459861u);
        const unsigned idx = h & (unsigned)(TSZ - 1);
        const float2 f = tl[idx];
        const float wgt = (ox ? wx : 1.0f - wx) * (oy ? wy : 1.0f - wy)
                                                * (oz ? wz : 1.0f - wz);
        a0 = fmaf(wgt, f.x, a0);
        a1 = fmaf(wgt, f.y, a1);
      }
      st[pt * 32 + 2 * l]     = (_Float16)a0;
      st[pt * 32 + 2 * l + 1] = (_Float16)a1;
    }
  }
  __syncthreads();

  // ---- Phase C: h = relu(enc @ sw0 + sb0)   [16,32]x[32,64] -> st[16][64]
  {
    const v16h a = load_A16(st, 32, 0, lane);
    v8f acc0 = {}, acc1 = {}, acc2 = {}, acc3 = {};
    acc0 = wmma16(a, load_Bfrag(wfrag, T_SW0(0), lane), acc0);
    acc1 = wmma16(a, load_Bfrag(wfrag, T_SW0(1), lane), acc1);
    acc2 = wmma16(a, load_Bfrag(wfrag, T_SW0(2), lane), acc2);
    acc3 = wmma16(a, load_Bfrag(wfrag, T_SW0(3), lane), acc3);
    store_relu(acc0, st, 64, 0, SB0, lane);
    store_relu(acc1, st, 64, 1, SB0, lane);
    store_relu(acc2, st, 64, 2, SB0, lane);
    store_relu(acc3, st, 64, 3, SB0, lane);
  }
  __syncthreads();

  // ---- Phase D: hv = h @ sw1 + sb1 -> sigma (n==0) + geo (n=1..15 -> c[0..14])
  {
    const v16h a0 = load_A16(st, 64, 0, lane);
    const v16h a1 = load_A16(st, 64, 1, lane);
    v8f acc = {};
    acc = wmma16(a0, load_Bfrag(wfrag, T_SW1(0), lane), acc);
    acc = wmma16(a1, load_Bfrag(wfrag, T_SW1(1), lane), acc);
    const int n = lane & 15, half = lane >> 4;
    const float bb = SB1[n];
#pragma unroll
    for (int j = 0; j < 8; ++j) {
      const int m = j + 8 * half;
      const float v = acc[j] + bb;
      if (n == 0) {
        const float t  = 100.0f * v;
        const float sp = (t > 20.0f) ? t : log1pf(expf(t));
        out[(size_t)(p0 + m) * 4 + 0] = sp * 0.01f;
      } else {
        st[m * 96 + (n - 1)] = (_Float16)v;   // geo -> c[0..14]
      }
    }
  }
  __syncthreads();

  // ---- Phase E: SH16 (lanes 0-15) + appearance row (lanes 16-31) -> c tile
  {
    const int pt = lane & 15;
    const int p  = p0 + pt;
    if (lane < 16) {
      const float dx = dvec[p * 3 + 0], dy = dvec[p * 3 + 1], dz = dvec[p * 3 + 2];
      const float inv = 1.0f / sqrtf(dx * dx + dy * dy + dz * dz);
      const float X = dx * inv, Y = dy * inv, Z = dz * inv;
      const float XX = X * X, YY = Y * Y, ZZ = Z * Z;
      const float XY = X * Y, YZ = Y * Z, XZ = X * Z;
      float sh[16];
      sh[0]  = 0.28209479177387814f;
      sh[1]  = -0.48860251190291987f * Y;
      sh[2]  =  0.48860251190291987f * Z;
      sh[3]  = -0.48860251190291987f * X;
      sh[4]  =  1.0925484305920792f * XY;
      sh[5]  = -1.0925484305920792f * YZ;
      sh[6]  =  0.94617469575756f * ZZ - 0.31539156525252005f;
      sh[7]  = -1.0925484305920792f * XZ;
      sh[8]  =  0.5462742152960396f * (XX - YY);
      sh[9]  = -0.5900435899266435f * Y * (3.0f * XX - YY);
      sh[10] =  2.890611442640554f * XY * Z;
      sh[11] = -0.4570457994644657f * Y * (4.0f * ZZ - XX - YY);
      sh[12] =  0.37317633259011546f * Z * (2.0f * ZZ - 3.0f * XX - 3.0f * YY);
      sh[13] = -0.4570457994644657f * X * (4.0f * ZZ - XX - YY);
      sh[14] =  1.445305721320277f * Z * (XX - YY);
      sh[15] = -0.5900435899266435f * X * (XX - 3.0f * YY);
#pragma unroll
      for (int i = 0; i < 16; ++i) st[pt * 96 + 15 + i] = (_Float16)sh[i];
    } else {
      const int ai = app_idx[p];
      const float4* ar4 = (const float4*)(atab + (size_t)ai * 48);
#pragma unroll
      for (int i4 = 0; i4 < 12; ++i4) {
        const float4 f = ar4[i4];
        _Float16* dst = st + pt * 96 + 31 + i4 * 4;
        dst[0] = (_Float16)f.x; dst[1] = (_Float16)f.y;
        dst[2] = (_Float16)f.z; dst[3] = (_Float16)f.w;
      }
#pragma unroll
      for (int i = 79; i < 96; ++i) st[pt * 96 + i] = (_Float16)0.0f; // pad
    }
  }
  __syncthreads();

  // ---- Phase F: c1 = relu(c @ cw0 + cb0)   [16,96]x[96,64]
  {
    v8f acc0 = {}, acc1 = {}, acc2 = {}, acc3 = {};
#pragma unroll
    for (int kc = 0; kc < 3; ++kc) {
      const v16h a = load_A16(st, 96, kc, lane);
      acc0 = wmma16(a, load_Bfrag(wfrag, T_CW0(kc, 0), lane), acc0);
      acc1 = wmma16(a, load_Bfrag(wfrag, T_CW0(kc, 1), lane), acc1);
      acc2 = wmma16(a, load_Bfrag(wfrag, T_CW0(kc, 2), lane), acc2);
      acc3 = wmma16(a, load_Bfrag(wfrag, T_CW0(kc, 3), lane), acc3);
    }
    store_relu(acc0, st, 64, 0, CB0, lane);
    store_relu(acc1, st, 64, 1, CB0, lane);
    store_relu(acc2, st, 64, 2, CB0, lane);
    store_relu(acc3, st, 64, 3, CB0, lane);
  }
  __syncthreads();

  // ---- Phase G: c2 = relu(c1 @ cw1 + cb1)   [16,64]x[64,64]
  {
    const v16h a0 = load_A16(st, 64, 0, lane);
    const v16h a1 = load_A16(st, 64, 1, lane);
    v8f acc0 = {}, acc1 = {}, acc2 = {}, acc3 = {};
    acc0 = wmma16(a0, load_Bfrag(wfrag, T_CW1(0, 0), lane), acc0);
    acc0 = wmma16(a1, load_Bfrag(wfrag, T_CW1(1, 0), lane), acc0);
    acc1 = wmma16(a0, load_Bfrag(wfrag, T_CW1(0, 1), lane), acc1);
    acc1 = wmma16(a1, load_Bfrag(wfrag, T_CW1(1, 1), lane), acc1);
    acc2 = wmma16(a0, load_Bfrag(wfrag, T_CW1(0, 2), lane), acc2);
    acc2 = wmma16(a1, load_Bfrag(wfrag, T_CW1(1, 2), lane), acc2);
    acc3 = wmma16(a0, load_Bfrag(wfrag, T_CW1(0, 3), lane), acc3);
    acc3 = wmma16(a1, load_Bfrag(wfrag, T_CW1(1, 3), lane), acc3);
    store_relu(acc0, st, 64, 0, CB1, lane);
    store_relu(acc1, st, 64, 1, CB1, lane);
    store_relu(acc2, st, 64, 2, CB1, lane);
    store_relu(acc3, st, 64, 3, CB1, lane);
  }
  __syncthreads();

  // ---- Phase H: rgb = sigmoid(c2 @ cw2 + cb2)   [16,64]x[64,16(3)]
  {
    const v16h a0 = load_A16(st, 64, 0, lane);
    const v16h a1 = load_A16(st, 64, 1, lane);
    v8f acc = {};
    acc = wmma16(a0, load_Bfrag(wfrag, T_CW2(0), lane), acc);
    acc = wmma16(a1, load_Bfrag(wfrag, T_CW2(1), lane), acc);
    const int n = lane & 15, half = lane >> 4;
    if (n < 3) {
      const float bb = CB2[n];
#pragma unroll
      for (int j = 0; j < 8; ++j) {
        const int m = j + 8 * half;
        const float v = acc[j] + bb;
        out[(size_t)(p0 + m) * 4 + 1 + n] = 1.0f / (1.0f + expf(-v));
      }
    }
  }
}

extern "C" void kernel_launch(void* const* d_in, const int* in_sizes, int n_in,
                              void* d_out, int out_size, void* d_ws, size_t ws_size,
                              hipStream_t stream) {
  const float* x    = (const float*)d_in[0];
  const float* dv   = (const float*)d_in[1];
  const int*   app  = (const int*)  d_in[2];
  const float* ht   = (const float*)d_in[3];
  const float* at   = (const float*)d_in[4];
  const float* sw0  = (const float*)d_in[5];
  const float* sb0  = (const float*)d_in[6];
  const float* sw1  = (const float*)d_in[7];
  const float* sb1  = (const float*)d_in[8];
  const float* cw0  = (const float*)d_in[9];
  const float* cb0  = (const float*)d_in[10];
  const float* cw1  = (const float*)d_in[11];
  const float* cb1  = (const float*)d_in[12];
  const float* cw2  = (const float*)d_in[13];
  const float* cb2  = (const float*)d_in[14];
  float* out = (float*)d_out;

  // RES levels computed in double to match numpy: floor(16 * b^l), b = (2048/16)^(1/15)
  ResT rt;
  const double b = exp(log(2048.0 / 16.0) / 15.0);
  for (int l = 0; l < 16; ++l) rt.r[l] = (float)floor(16.0 * pow(b, (double)l));

  const int n = in_sizes[0] / 3;          // 2,097,152 points
  dim3 grid((unsigned)(n / 128)), block(256);
  hipLaunchKernelGGL(nerf_fused, grid, block, 0, stream,
                     x, dv, app, ht, at, sw0, sb0, sw1, sb1,
                     cw0, cb0, cw1, cb1, cw2, cb2, out, rt);
}
